// EKV_ResNet_28716151341203
// MI455X (gfx1250) — matrix operand order repaired and checked by reading, not compiled, not run on hardware
//
#include <hip/hip_runtime.h>
#include <hip/hip_bf16.h>

#define EKV_ALPHA 0.0005625f
#define EKV_S     13.333333333333334f   // 1/(2*1.5*0.025)
#define EKV_VD    0.1f
#define R_TIA     10000.0f
#define VMAXV     9.0f
#define BN_EPS    1e-5f

typedef __attribute__((ext_vector_type(16))) _Float16 v16h;
typedef __attribute__((ext_vector_type(8)))  float    v8f;

__device__ __forceinline__ float softplus_f(float x) {
    // stable: max(x,0) + log1p(exp(-|x|)); exp arg <= 0, never overflows
    return fmaxf(x, 0.0f) + __logf(1.0f + __expf(-fabsf(x)));
}

// ---------------------------------------------------------------------------
// EKV conv: out[b,o,ho,wo] = R_TIA*ALPHA * sum_{c,i,j} f(V[b,c,hi,wi]*scale - th[o,c,i,j])
// Templated on K so the 3x3 tap loops fully unroll: 18 independent
// exp/log chains per channel iteration keep the trans pipe saturated.
// Accumulates per-channel sum / sumsq for training-mode BN via LDS
// reduction + one global float atomic per workgroup.
// blockDim.x == 256, grid = (tiles over B*Ho*Wo, O)
// ---------------------------------------------------------------------------
template <int K>
__global__ void ekv_conv_kernel(const float* __restrict__ Vin,
                                const float* __restrict__ theta,
                                float* __restrict__ out,
                                float* __restrict__ stats,
                                int B, int C, int H, int W, int O,
                                int stride, int pad, int Ho, int Wo,
                                float in_scale) {
    __shared__ float sTheta[288];   // max CKK = 32*3*3
    __shared__ float sRed[512];
    const int tid = threadIdx.x;
    const int o   = blockIdx.y;
    const int CKK = C * K * K;
    for (int t = tid; t < CKK; t += 256)
        sTheta[t] = theta[o * CKK + t];
    __syncthreads();

    const int L   = Ho * Wo;
    const int idx = blockIdx.x * 256 + tid;
    float val = 0.0f, valsq = 0.0f;
    if (idx < B * L) {
        const int b   = idx / L;
        const int rem = idx - b * L;
        const int ho  = rem / Wo;
        const int wo  = rem - ho * Wo;
        const int hbase = ho * stride - pad;
        const int wbase = wo * stride - pad;

        // gather the K*K input taps once per channel, then evaluate all
        // 2*K*K softplus chains with full ILP
        float acc = 0.0f;
        for (int c = 0; c < C; ++c) {
            const float* Vc = Vin + (size_t)(b * C + c) * H * W;
            const float* th = sTheta + c * K * K;
            __builtin_prefetch(Vc + H * W, 0, 0);   // global_prefetch_b8: next channel
            float v[K * K];
#pragma unroll
            for (int i = 0; i < K; ++i) {
                const int  hi  = hbase + i;
                const bool hok = (hi >= 0) && (hi < H);
#pragma unroll
                for (int j = 0; j < K; ++j) {
                    const int wi = wbase + j;
                    // padded positions contribute f(0 - theta) (nonzero!)
                    v[i * K + j] = (hok && wi >= 0 && wi < W)
                                       ? Vc[hi * W + wi] * in_scale : 0.0f;
                }
            }
#pragma unroll
            for (int t = 0; t < K * K; ++t) {
                float d  = v[t] - th[t];
                float t1 = softplus_f(d * EKV_S);
                float t2 = softplus_f((d - EKV_VD) * EKV_S);
                acc = fmaf(t1, t1, acc);
                acc = fmaf(-t2, t2, acc);
            }
        }
        val = acc * (EKV_ALPHA * R_TIA);
        out[(size_t)(b * O + o) * L + rem] = val;
        valsq = val * val;
    }
    // BN statistics: LDS tree reduce, one atomic per workgroup per channel
    sRed[tid]       = val;
    sRed[256 + tid] = valsq;
    __syncthreads();
    for (int s = 128; s > 0; s >>= 1) {
        if (tid < s) {
            sRed[tid]       += sRed[tid + s];
            sRed[256 + tid] += sRed[256 + tid + s];
        }
        __syncthreads();
    }
    if (tid == 0) {
        atomicAdd(&stats[o],     sRed[0]);
        atomicAdd(&stats[O + o], sRed[256]);
    }
}

// mode bit0: clip to [0, VMAX]; bit1: add residual
__global__ void bn_apply_kernel(const float* __restrict__ x,
                                const float* __restrict__ stats,
                                const float* __restrict__ gw,
                                const float* __restrict__ gb,
                                const float* __restrict__ addp,
                                float* __restrict__ out,
                                int C, int HW, int total, int mode) {
    int idx = blockIdx.x * blockDim.x + threadIdx.x;
    if (idx >= total) return;
    int   c    = (idx / HW) % C;
    float n    = (float)(total / C);
    float mean = stats[c] / n;
    float var  = stats[C + c] / n - mean * mean;   // biased variance (matches ref)
    float y = (x[idx] - mean) * rsqrtf(var + BN_EPS) * gw[c] + gb[c];
    if (mode & 2) y += addp[idx];
    if (mode & 1) y = fminf(fmaxf(y, 0.0f), VMAXV);
    out[idx] = y;
}

__global__ void zero_kernel(float* __restrict__ p, int n) {
    int i = blockIdx.x * blockDim.x + threadIdx.x;
    if (i < n) p[i] = 0.0f;
}

__global__ void avgpool_kernel(const float* __restrict__ V,
                               float* __restrict__ feat, int BC, int HW) {
    int idx = blockIdx.x * blockDim.x + threadIdx.x;   // b*C + c
    if (idx >= BC) return;
    const float* p = V + (size_t)idx * HW;
    float s = 0.0f;
    for (int i = 0; i < HW; ++i) s += p[i];
    feat[idx] = s / (float)HW;
}

// ---------------------------------------------------------------------------
// FC via WMMA: out[64,10] = feat[64,32] @ fc_w[10,32]^T + fc_b
// M=64 -> 4 tiles of 16, K=32 (one v_wmma_f32_16x16x32_f16), N=10 padded to 16.
// One wave (32 lanes) per block; EXEC all-ones at the WMMA (selects, no branches).
// ---------------------------------------------------------------------------
__global__ void fc_wmma_kernel(const float* __restrict__ feat,
                               const float* __restrict__ fc_w,
                               const float* __restrict__ fc_b,
                               float* __restrict__ out) {
    const int lane = threadIdx.x;     // 0..31
    const int tile = blockIdx.x;      // 0..3 (M tiles)
    const int half = lane >> 4;
    const int mn   = lane & 15;       // M for A, N for B/D
    const float bvalid = (mn < 10) ? 1.0f : 0.0f;
    const int   brow   = (mn < 10) ? mn : 0;

    v16h a, bm;
#pragma unroll
    for (int e = 0; e < 16; ++e) {
        // A fragment (16x32 f16): VGPR e/2 halfword e%2 -> K = e (+8 if e>=8) + half*8
        int ka = e + ((e >= 8) ? 8 : 0) + half * 8;
        a[e] = (_Float16)feat[(tile * 16 + mn) * 32 + ka];
        // B fragment (32x16 f16): K = e (+8 if e>=8) + half*8, N = lane%16 (zero-padded N>=10)
        int kb = e + ((e >= 8) ? 8 : 0) + half * 8;
        bm[e] = (_Float16)(fc_w[brow * 32 + kb] * bvalid);
    }
    v8f c = {};
    c = __builtin_amdgcn_wmma_f32_16x16x32_f16(false, a, false, bm,
                                               (short)0, c, false, false);
    if (mn < 10) {
        float bias = fc_b[mn];
#pragma unroll
        for (int r = 0; r < 8; ++r) {
            int m = tile * 16 + r + 8 * half;   // D: VGPR r -> M = r + 8*half
            out[m * 10 + mn] = c[r] + bias;
        }
    }
}

// ---------------------------------------------------------------------------
extern "C" void kernel_launch(void* const* d_in, const int* in_sizes, int n_in,
                              void* d_out, int out_size, void* d_ws, size_t ws_size,
                              hipStream_t stream) {
    (void)in_sizes; (void)n_in; (void)out_size; (void)ws_size;
    const int B = 64;
    const float* x           = (const float*)d_in[0];
    const float* conv1_theta = (const float*)d_in[1];
    const float* bn1_w       = (const float*)d_in[2];
    const float* bn1_b       = (const float*)d_in[3];
    int cursor = 4;

    // workspace layout (floats)
    const size_t BUF = 64 * 8 * 32 * 32;  // 524288 (largest activation)
    float* w     = (float*)d_ws;
    float* cur   = w;
    float* raw1  = w + BUF;
    float* act1  = w + 2 * BUF;
    float* raw2  = w + 3 * BUF;
    float* scb   = w + 4 * BUF;
    float* stats = w + 5 * BUF;           // 3 slices of 64 (sum|sumsq per conv)
    float* feat  = stats + 256;           // [64,32]

    auto conv = [&](const float* in, const float* th, float* outp, float* st,
                    int C, int H, int O, int K, int s, int pad, float in_scale) {
        int Ho = (H + 2 * pad - K) / s + 1;
        int L  = Ho * Ho;
        dim3 grid((B * L + 255) / 256, O);
        if (K == 3)
            ekv_conv_kernel<3><<<grid, 256, 0, stream>>>(
                in, th, outp, st, B, C, H, H, O, s, pad, Ho, Ho, in_scale);
        else
            ekv_conv_kernel<1><<<grid, 256, 0, stream>>>(
                in, th, outp, st, B, C, H, H, O, s, pad, Ho, Ho, in_scale);
    };
    auto bn = [&](const float* in, const float* st, const float* gw, const float* gb,
                  const float* addp, float* outp, int C, int L, int mode) {
        int total = B * C * L;
        bn_apply_kernel<<<(total + 255) / 256, 256, 0, stream>>>(
            in, st, gw, gb, addp, outp, C, L, total, mode);
    };
    auto zero = [&](float* p, int n) {
        zero_kernel<<<(n + 255) / 256, 256, 0, stream>>>(p, n);
    };

    // stem: V = x*9 folded into conv via in_scale; bn + clip
    zero(stats, 64);
    conv(x, conv1_theta, raw1, stats, 3, 32, 8, 3, 1, 1, VMAXV);
    bn(raw1, stats, bn1_w, bn1_b, nullptr, cur, 8, 32 * 32, 1);

    static const int cfg[9][3] = {{8,8,1},{8,8,1},{8,8,1},
                                  {8,16,2},{16,16,1},{16,16,1},
                                  {16,32,2},{32,32,1},{32,32,1}};
    int H = 32;
    for (int bi = 0; bi < 9; ++bi) {
        const int cin = cfg[bi][0], cout = cfg[bi][1], s = cfg[bi][2];
        const bool has_sc = (s != 1) || (cin != cout);
        const float* th1 = (const float*)d_in[cursor++];
        const float* w1  = (const float*)d_in[cursor++];
        const float* b1  = (const float*)d_in[cursor++];
        const float* th2 = (const float*)d_in[cursor++];
        const float* w2  = (const float*)d_in[cursor++];
        const float* b2  = (const float*)d_in[cursor++];
        const float *thsc = nullptr, *wsc = nullptr, *bsc = nullptr;
        if (has_sc) {
            thsc = (const float*)d_in[cursor++];
            wsc  = (const float*)d_in[cursor++];
            bsc  = (const float*)d_in[cursor++];
        }
        const int Ho = H / s;
        const int L  = Ho * Ho;

        zero(stats, 192);
        conv(cur, th1, raw1, stats, cin, H, cout, 3, s, 1, 1.0f);
        bn(raw1, stats, w1, b1, nullptr, act1, cout, L, 1);       // clip
        conv(act1, th2, raw2, stats + 64, cout, Ho, cout, 3, 1, 1, 1.0f);
        const float* addsrc;
        if (has_sc) {
            conv(cur, thsc, scb, stats + 128, cin, H, cout, 1, s, 0, 1.0f);
            bn(scb, stats + 128, wsc, bsc, nullptr, scb, cout, L, 0);  // no clip
            addsrc = scb;
        } else {
            addsrc = cur;       // identity shortcut
        }
        bn(raw2, stats + 64, w2, b2, addsrc, cur, cout, L, 3);    // add + clip -> cur
        H = Ho;
    }

    // adaptive avg pool (8x8 -> 1) and WMMA FC head
    avgpool_kernel<<<(64 * 32 + 255) / 256, 256, 0, stream>>>(cur, feat, 64 * 32, 64);
    const float* fc_w = (const float*)d_in[cursor++];
    const float* fc_b = (const float*)d_in[cursor++];
    fc_wmma_kernel<<<4, 32, 0, stream>>>(feat, fc_w, fc_b, (float*)d_out);
}
